// DeformableModulatedConv2d_74543452390133
// MI455X (gfx1250) — matrix operand order repaired and checked by reading, not compiled, not run on hardware
//
#include <hip/hip_runtime.h>

// ModulatedConv2d for MI455X (gfx1250, wave32, WMMA bf16 + async global->LDS).
// Phase 0: style projection s[b][ic]
// Phase 1: demod + bf16 weight pack, K ordered as (r, ic):  wbf[b][oc][r*512+ic]
// Phase 2: x NCHW f32 -> NHWC bf16 transpose (xh[b][y][x][ic])
// Phase 3: per-batch implicit-GEMM conv, double-buffered LDS fed by
//          GLOBAL_LOAD_ASYNC_TO_LDS_B128, compute via V_WMMA_F32_16X16X32_BF16

#define IN_CH     512
#define OUT_CH    512
#define STYLE_DIM 512
#define BATCH     16
#define HW        64
#define NPIX      4096              // HW*HW
#define KTOT      4608              // 9 * IN_CH
#define SCALE_C   0.014731391274719739f   // 1/sqrt(4608)

#define BM 128
#define BN 128
#define BK 32

typedef __attribute__((ext_vector_type(16))) __bf16 v16bf;
typedef __attribute__((ext_vector_type(8)))  __bf16 v8bf;
typedef __attribute__((ext_vector_type(8)))  float  v8f;

__device__ __forceinline__ unsigned short f2bf(float f) {
  union { float f; unsigned u; } x; x.f = f;
  unsigned r = x.u + 0x7FFFu + ((x.u >> 16) & 1u);   // round-to-nearest-even
  return (unsigned short)(r >> 16);
}

// --- CDNA5 async global->LDS (ASYNCcnt), CDNA5 ISA ---
// LDS offset = low 32 bits of flat pointer (flat->LDS maps addr[31:0]).
__device__ __forceinline__ void async_ld_b128(const void* gaddr, void* lds) {
  unsigned l = (unsigned)(unsigned long long)lds;
  asm volatile("global_load_async_to_lds_b128 %0, %1, off"
               :: "v"(l), "v"(gaddr) : "memory");
}
__device__ __forceinline__ void wait_async0() {
  asm volatile("s_wait_asynccnt 0" ::: "memory");
}

// ---------------- Phase 0: s = style @ mod_w^T + mod_b ----------------
__global__ __launch_bounds__(256)
void style_proj_kernel(const float* __restrict__ style,
                       const float* __restrict__ mod_w,
                       const float* __restrict__ mod_b,
                       float* __restrict__ s) {
  int idx = blockIdx.x * 256 + threadIdx.x;
  int b  = idx >> 9;
  int ic = idx & 511;
  const float* st = style + (size_t)b * STYLE_DIM;
  const float* mw = mod_w + (size_t)ic * STYLE_DIM;
  float acc = 0.f;
  #pragma unroll 8
  for (int j = 0; j < STYLE_DIM; ++j) acc += st[j] * mw[j];
  s[idx] = acc + mod_b[ic];
}

// ---------------- Phase 1: demod + bf16 weight pack (r-major K) ----------------
__global__ __launch_bounds__(256)
void modulate_demod_kernel(const float* __restrict__ weight,
                           const float* __restrict__ s,
                           unsigned short* __restrict__ wbf) {
  const int oc  = blockIdx.x;
  const int b   = blockIdx.y;
  const int tid = threadIdx.x;
  __shared__ float red[256];
  __shared__ float demod_sh;

  float partial = 0.f;
  for (int ic = tid; ic < IN_CH; ic += 256) {
    float sv = s[b * IN_CH + ic] * SCALE_C;
    const float* wp = weight + ((size_t)oc * IN_CH + ic) * 9;
    #pragma unroll
    for (int r = 0; r < 9; ++r) { float v = wp[r] * sv; partial += v * v; }
  }
  red[tid] = partial;
  __syncthreads();
  #pragma unroll
  for (int off = 128; off > 0; off >>= 1) {
    if (tid < off) red[tid] += red[tid + off];
    __syncthreads();
  }
  if (tid == 0) demod_sh = rsqrtf(red[0] + 1e-8f);
  __syncthreads();
  const float dm = demod_sh;

  unsigned short* dst = wbf + ((size_t)b * OUT_CH + oc) * KTOT;
  for (int ic = tid; ic < IN_CH; ic += 256) {
    float m = s[b * IN_CH + ic] * SCALE_C * dm;
    const float* wp = weight + ((size_t)oc * IN_CH + ic) * 9;
    #pragma unroll
    for (int r = 0; r < 9; ++r) dst[r * IN_CH + ic] = f2bf(wp[r] * m);  // (r, ic)
  }
}

// ---------------- Phase 2: x NCHW f32 -> NHWC bf16 ----------------
__global__ __launch_bounds__(256)
void x_to_nhwc_kernel(const float* __restrict__ x, unsigned short* __restrict__ xh) {
  const int y    = blockIdx.x;
  const int b    = blockIdx.y;
  const int xcol = threadIdx.x & 63;
  const int icg  = threadIdx.x >> 6;          // 0..3, 128 ic each
  const float* src = x + (((size_t)b * IN_CH + icg * 128) * HW + y) * HW + xcol;
  unsigned short* dst = xh + ((size_t)b * NPIX + y * HW + xcol) * IN_CH + icg * 128;
  #pragma unroll 4
  for (int g = 0; g < 16; ++g) {              // 8 ic per pack -> one b128 store
    unsigned short h[8];
    #pragma unroll
    for (int e = 0; e < 8; ++e) h[e] = f2bf(src[(size_t)(g * 8 + e) * NPIX]);
    uint4 pk;
    pk.x = (unsigned)h[0] | ((unsigned)h[1] << 16);
    pk.y = (unsigned)h[2] | ((unsigned)h[3] << 16);
    pk.z = (unsigned)h[4] | ((unsigned)h[5] << 16);
    pk.w = (unsigned)h[6] | ((unsigned)h[7] << 16);
    *(uint4*)(dst + g * 8) = pk;
  }
}

// ---------------- Phase 3: implicit-GEMM conv with WMMA ----------------
// Grid: (32 pixel-blocks, 4 oc-blocks, 16 batches). Block = 256 threads = 8 waves.
// Wave grid 4(M) x 2(N); each wave: 32 oc x 64 pix = 2x4 WMMA tiles.
__global__ __launch_bounds__(256)
void modconv_gemm_kernel(const unsigned short* __restrict__ xh,
                         const unsigned short* __restrict__ wbf,
                         float* __restrict__ out) {
  __shared__ __align__(32) unsigned short lA[2][BM * BK];  // [oc][k]
  __shared__ __align__(32) unsigned short lB[2][BN * BK];  // [pix][k]

  const int tid     = threadIdx.x;
  const int b       = blockIdx.z;
  const int ocBase  = blockIdx.y * BM;
  const int pBase   = blockIdx.x * BN;
  const int pixRow0 = blockIdx.x * 2;       // 2 image rows per pixel-block

  const int lane  = tid & 31;
  const int wave  = tid >> 5;
  const int mWave = wave >> 1;              // 0..3
  const int nWave = wave & 1;               // 0..1
  const int half  = lane >> 4;              // 0..1
  const int l16   = lane & 15;

  // A mapping: 2 threads per oc row, 16 elems (32B) each
  const int aRow = tid >> 1;
  const int aSeg = tid & 1;
  const unsigned short* aGlob =
      wbf + ((size_t)b * OUT_CH + ocBase + aRow) * KTOT + aSeg * 16;

  // B mapping: 2 threads per pixel, 16 ic (32B) each
  const int bPix = tid >> 1;                // 0..127
  const int bSeg = tid & 1;
  const int prow = bPix >> 6;               // 0..1
  const int gx   = bPix & 63;
  const unsigned short* xBase = xh + (size_t)b * NPIX * IN_CH + bSeg * 16;

  // Issue async copies for one K-chunk into LDS buffer `bsel`.
  auto stage = [&](int k0, int bsel) {
    unsigned short* la = &lA[bsel][aRow * BK + aSeg * 16];
    async_ld_b128(aGlob + k0,     la);
    async_ld_b128(aGlob + k0 + 8, la + 8);

    int r   = k0 >> 9;                      // uniform tap index (SALU)
    int icb = k0 & 511;
    int dy  = r / 3 - 1;
    int dx  = (r - (r / 3) * 3) - 1;
    int y   = pixRow0 + prow + dy;
    int xx  = gx + dx;
    unsigned short* lb = &lB[bsel][bPix * BK + bSeg * 16];
    if (((unsigned)y < (unsigned)HW) & ((unsigned)xx < (unsigned)HW)) {
      const unsigned short* g = xBase + ((size_t)y * HW + xx) * IN_CH + icb;
      async_ld_b128(g,     lb);
      async_ld_b128(g + 8, lb + 8);
    } else {                                // zero-fill halo (image-border taps, any block)
      uint4 z = make_uint4(0, 0, 0, 0);
      *(uint4*)lb = z;
      *(uint4*)(lb + 8) = z;
    }
  };

  v8f acc[2][4];
  #pragma unroll
  for (int mt = 0; mt < 2; ++mt)
    #pragma unroll
    for (int nt = 0; nt < 4; ++nt)
      acc[mt][nt] = (v8f){0.f,0.f,0.f,0.f,0.f,0.f,0.f,0.f};

  stage(0, 0);
  int buf = 0;
  for (int k0 = 0; k0 < KTOT; k0 += BK) {
    wait_async0();        // my async writes to lds[buf] landed
    __syncthreads();      // everyone's landed; prior reads of lds[buf^1] retired

    if (k0 + BK < KTOT) stage(k0 + BK, buf ^ 1);   // overlaps WMMA below

    // fragment loads (ISA §7.12.2 layouts) + WMMA
    union V16 { v16bf v; v8bf h[2]; };
    v16bf aF[2], bF[4];
    #pragma unroll
    for (int mt = 0; mt < 2; ++mt) {
      int row = mWave * 32 + mt * 16 + l16;
      V16 t;
      t.h[0] = *(const v8bf*)&lA[buf][row * BK + half * 8];       // K = h*8..h*8+7
      t.h[1] = *(const v8bf*)&lA[buf][row * BK + 16 + half * 8];  // K = 16+h*8..
      aF[mt] = t.v;
    }
    #pragma unroll
    for (int nt = 0; nt < 4; ++nt) {
      int col = nWave * 64 + nt * 16 + l16;
      V16 t;
      t.h[0] = *(const v8bf*)&lB[buf][col * BK + half * 8];       // K = h*8..h*8+7
      t.h[1] = *(const v8bf*)&lB[buf][col * BK + 16 + half * 8];  // K = 16+h*8..
      bF[nt] = t.v;
    }
    #pragma unroll
    for (int mt = 0; mt < 2; ++mt)
      #pragma unroll
      for (int nt = 0; nt < 4; ++nt)
        acc[mt][nt] = __builtin_amdgcn_wmma_f32_16x16x32_bf16(
            false, aF[mt], false, bF[nt], (short)0, acc[mt][nt], false, false);

    buf ^= 1;
  }

  // epilogue: C layout VGPR i -> M = half*8+i, N = l16
  #pragma unroll
  for (int mt = 0; mt < 2; ++mt) {
    #pragma unroll
    for (int nt = 0; nt < 4; ++nt) {
      int p = pBase + nWave * 64 + nt * 16 + l16;
      #pragma unroll
      for (int i = 0; i < 8; ++i) {
        int oc = ocBase + mWave * 32 + mt * 16 + half * 8 + i;
        out[((size_t)b * OUT_CH + oc) * NPIX + p] = acc[mt][nt][i];
      }
    }
  }
}

extern "C" void kernel_launch(void* const* d_in, const int* in_sizes, int n_in,
                              void* d_out, int out_size, void* d_ws, size_t ws_size,
                              hipStream_t stream) {
  (void)in_sizes; (void)n_in; (void)out_size; (void)ws_size;
  const float* x      = (const float*)d_in[0];
  const float* style  = (const float*)d_in[1];
  const float* weight = (const float*)d_in[2];
  const float* mod_w  = (const float*)d_in[3];
  const float* mod_b  = (const float*)d_in[4];
  float* out = (float*)d_out;

  // workspace layout:
  //   s   : 16x512 f32            =     32 KB   @ 0
  //   wbf : 16x512x4608 bf16      = 75.50 MB    @ 32768
  //   xh  : 16x64x64x512 bf16     = 67.11 MB    @ 32768 + 75497472
  float* s = (float*)d_ws;
  unsigned short* wbf = (unsigned short*)((char*)d_ws + 32768);
  unsigned short* xh  = (unsigned short*)((char*)d_ws + 32768 + (size_t)BATCH * OUT_CH * KTOT * 2);

  style_proj_kernel<<<dim3((BATCH * IN_CH) / 256), 256, 0, stream>>>(
      style, mod_w, mod_b, s);
  modulate_demod_kernel<<<dim3(OUT_CH, BATCH), 256, 0, stream>>>(
      weight, s, wbf);
  x_to_nhwc_kernel<<<dim3(HW, BATCH), 256, 0, stream>>>(x, xh);
  modconv_gemm_kernel<<<dim3(NPIX / BN, OUT_CH / BM, BATCH), 256, 0, stream>>>(
      xh, wbf, out);
}
